// BagleyAttention_8761733284178
// MI455X (gfx1250) — matrix operand order repaired and checked by reading, compiled, not run on hardware
//
#include <hip/hip_runtime.h>
#include <hip/hip_bf16.h>

typedef __attribute__((ext_vector_type(16))) __bf16 v16bf;
typedef __attribute__((ext_vector_type(8)))  float  v8f;
typedef unsigned short ushort_t;
typedef unsigned int   uint_t;

union Frag16 { v16bf v; uint_t u[8]; };

static constexpr int S  = 2048;
static constexpr int D  = 4096;
static constexpr int H  = 32;
static constexpr int KV = 8;
static constexpr int Dh = 128;

__device__ __forceinline__ ushort_t f2bf1(float f) {
    uint_t u = __float_as_uint(f);
    return (ushort_t)((u + 0x7FFFu + ((u >> 16) & 1u)) >> 16);
}
__device__ __forceinline__ uint_t pack2(float a, float b) {
    return (uint_t)f2bf1(a) | ((uint_t)f2bf1(b) << 16);
}

// Async 16B global -> LDS copy (CDNA5, ASYNCcnt-tracked). saddr + 32b voffset.
__device__ __forceinline__ void async_b128(unsigned lds_off, unsigned voff,
                                           const void* base) {
    asm volatile("global_load_async_to_lds_b128 %0, %1, %2"
                 :: "v"(lds_off), "v"(voff), "s"(base) : "memory");
}
__device__ __forceinline__ void wait_async0() {
    asm volatile("s_wait_asynccnt 0x0" ::: "memory");
}

// ---------------------------------------------------------------------------
// fp32 -> bf16 bulk conversion (8 elements / thread)
// ---------------------------------------------------------------------------
__global__ __launch_bounds__(256)
void f32_to_bf16(const float* __restrict__ src, ushort_t* __restrict__ dst, int n8) {
    int i = blockIdx.x * 256 + threadIdx.x;
    if (i < n8) {
        float4 f0 = ((const float4*)src)[(size_t)i * 2];
        float4 f1 = ((const float4*)src)[(size_t)i * 2 + 1];
        uint4 o;
        o.x = pack2(f0.x, f0.y); o.y = pack2(f0.z, f0.w);
        o.z = pack2(f1.x, f1.y); o.w = pack2(f1.z, f1.w);
        ((uint4*)dst)[i] = o;
    }
}

// ---------------------------------------------------------------------------
// Pure-bf16 GEMM: C[M,N](fp32) = A[M,K](bf16) * W[N,K](bf16)^T
// 256 thr = 8 waves (2x4), block tile 128x128, wave tile 64x32, K-step 64.
// Double-buffered LDS filled by global_load_async_to_lds_b128.
// ---------------------------------------------------------------------------
__global__ __launch_bounds__(256)
void gemm_wmma_async(const ushort_t* __restrict__ A, const ushort_t* __restrict__ W,
                     float* __restrict__ C, int M, int N, int K) {
    __shared__ __align__(16) ushort_t As[2][128 * 64];   // [m][k]
    __shared__ __align__(16) ushort_t Bs[2][128 * 64];   // [n][k]

    const int tid  = threadIdx.x;
    const int lane = tid & 31;
    const int wave = tid >> 5;
    const int half = lane >> 4;
    const int l16  = lane & 15;
    const int wm   = wave >> 2;
    const int wn   = wave & 3;
    const int m0   = blockIdx.y * 128;
    const int n0   = blockIdx.x * 128;

    auto issue = [&](int st, int buf) {
        const int k0 = st << 6;
        // 128 rows x 64 bf16 = 1024 16B chunks per tile; 4 per thread each
#pragma unroll
        for (int i = 0; i < 4; ++i) {
            int e = tid + i * 256;
            int row = e >> 3, seg = e & 7;
            unsigned lds = (unsigned)(size_t)&As[buf][row * 64 + seg * 8];
            unsigned vof = (unsigned)(((size_t)(m0 + row) * K + k0 + seg * 8) * 2);
            async_b128(lds, vof, A);
        }
#pragma unroll
        for (int i = 0; i < 4; ++i) {
            int e = tid + i * 256;
            int row = e >> 3, seg = e & 7;
            unsigned lds = (unsigned)(size_t)&Bs[buf][row * 64 + seg * 8];
            unsigned vof = (unsigned)(((size_t)(n0 + row) * K + k0 + seg * 8) * 2);
            async_b128(lds, vof, W);
        }
    };

    v8f zero = {};
    v8f acc[4][2];
#pragma unroll
    for (int i = 0; i < 4; ++i)
#pragma unroll
        for (int j = 0; j < 2; ++j) acc[i][j] = zero;

    const int steps = K >> 6;
    issue(0, 0);
    for (int st = 0; st < steps; ++st) {
        wait_async0();
        __syncthreads();
        if (st + 1 < steps) issue(st + 1, (st + 1) & 1);

        const uint_t* Asu = (const uint_t*)As[st & 1];
        const uint_t* Bsu = (const uint_t*)Bs[st & 1];
#pragma unroll
        for (int kk = 0; kk < 2; ++kk) {       // two K=32 sub-steps
            const int kb = kk * 16;            // uint offset within 64-wide row
            Frag16 a[4], b[2];
#pragma unroll
            for (int i = 0; i < 4; ++i) {
                int row = wm * 64 + i * 16 + l16;
#pragma unroll
                for (int j = 0; j < 8; ++j)
                    a[i].u[j] = Asu[row * 32 + kb + ((j < 4) ? j : j + 4) + 4 * half];
            }
#pragma unroll
            for (int i = 0; i < 2; ++i) {
                int col = wn * 32 + i * 16 + l16;
#pragma unroll
                for (int j = 0; j < 8; ++j)
                    b[i].u[j] = Bsu[col * 32 + kb + ((j < 4) ? j : j + 4) + 4 * half];
            }
#pragma unroll
            for (int i = 0; i < 4; ++i)
#pragma unroll
                for (int j = 0; j < 2; ++j)
                    acc[i][j] = __builtin_amdgcn_wmma_f32_16x16x32_bf16(
                        false, a[i].v, false, b[j].v, (short)0, acc[i][j], false, false);
        }
        __syncthreads();
    }

#pragma unroll
    for (int i = 0; i < 4; ++i)
#pragma unroll
        for (int j = 0; j < 2; ++j)
#pragma unroll
            for (int r = 0; r < 8; ++r) {
                int row = m0 + wm * 64 + i * 16 + r + 8 * half;
                int col = n0 + wn * 32 + j * 16 + l16;
                C[(size_t)row * N + col] = acc[i][j][r];
            }
}

// ---------------------------------------------------------------------------
// YaRN RoPE on Q (with 1/sqrt(Dh) folded in) and K; bf16 repack head-major.
// V repacked bf16 block-transposed: Vb[kv][s/32][d][s%32] so the attention
// kernel can stage it with contiguous async copies.
// ---------------------------------------------------------------------------
__global__ __launch_bounds__(256)
void rope_convert(const float* __restrict__ Qf, const float* __restrict__ Kf,
                  const float* __restrict__ Vf, const float* __restrict__ cosb,
                  const float* __restrict__ sinb, ushort_t* __restrict__ Qr,
                  ushort_t* __restrict__ Kr, ushort_t* __restrict__ Vb) {
    const float QSCALE = 0.08838834764831845f;   // 1/sqrt(128)
    const size_t NQ = (size_t)H * S * 64;
    const size_t NK = (size_t)KV * S * 64;
    const size_t NV = (size_t)KV * S * Dh;
    size_t gid = (size_t)blockIdx.x * 256 + threadIdx.x;
    if (gid < NQ) {
        int h = (int)(gid / ((size_t)S * 64));
        int rem = (int)(gid % ((size_t)S * 64));
        int s = rem >> 6, p = rem & 63;
        float q1 = Qf[(size_t)s * D + h * Dh + p];
        float q2 = Qf[(size_t)s * D + h * Dh + p + 64];
        float c1 = cosb[s * Dh + p],      s1 = sinb[s * Dh + p];
        float c2 = cosb[s * Dh + p + 64], s2 = sinb[s * Dh + p + 64];
        size_t o = ((size_t)h * S + s) * Dh + p;
        Qr[o]      = f2bf1((q1 * c1 - q2 * s1) * QSCALE);
        Qr[o + 64] = f2bf1((q2 * c2 + q1 * s2) * QSCALE);
    } else if (gid < NQ + NK) {
        size_t g = gid - NQ;
        int kv = (int)(g / ((size_t)S * 64));
        int rem = (int)(g % ((size_t)S * 64));
        int s = rem >> 6, p = rem & 63;
        float k1 = Kf[(size_t)s * (KV * Dh) + kv * Dh + p];
        float k2 = Kf[(size_t)s * (KV * Dh) + kv * Dh + p + 64];
        float c1 = cosb[s * Dh + p],      s1 = sinb[s * Dh + p];
        float c2 = cosb[s * Dh + p + 64], s2 = sinb[s * Dh + p + 64];
        size_t o = ((size_t)kv * S + s) * Dh + p;
        Kr[o]      = f2bf1(k1 * c1 - k2 * s1);
        Kr[o + 64] = f2bf1(k2 * c2 + k1 * s2);
    } else if (gid < NQ + NK + NV) {
        size_t g = gid - NQ - NK;
        int kv = (int)(g / ((size_t)S * Dh));
        int rem = (int)(g % ((size_t)S * Dh));
        int s = rem >> 7, d = rem & 127;
        size_t dst = ((size_t)(kv * (S / 32) + (s >> 5)) * Dh + d) * 32 + (s & 31);
        Vb[dst] = f2bf1(Vf[(size_t)s * (KV * Dh) + kv * Dh + d]);
    }
}

// ---------------------------------------------------------------------------
// Flash attention: grid (S/64, H), 128 thr = 4 waves; each wave owns 16 query
// rows. K/V blocks double-buffered via async-to-LDS; QK^T and P*V in WMMA;
// fp32 online softmax with shfl_xor row reductions.
// ---------------------------------------------------------------------------
__global__ __launch_bounds__(128)
void flash_attn(const ushort_t* __restrict__ Qr, const ushort_t* __restrict__ Kr,
                const ushort_t* __restrict__ Vb, ushort_t* __restrict__ AO) {
    __shared__ __align__(16) ushort_t Kbuf[2][32 * 128];   // [key][d]
    __shared__ __align__(16) ushort_t Vbuf[2][128 * 32];   // [d][key] (pre-transposed)
    __shared__ __align__(16) ushort_t Ps[4][16 * 32];      // per-wave probs [m][key]

    const int tid  = threadIdx.x;
    const int lane = tid & 31;
    const int wave = tid >> 5;
    const int half = lane >> 4;
    const int l16  = lane & 15;
    const int h    = blockIdx.y;
    const int kv   = h >> 2;                 // G = H/KV = 4
    const int q0   = blockIdx.x * 64;
    const int qr0  = q0 + wave * 16;

    auto issueKV = [&](int kb, int buf) {
        const ushort_t* Kgp = Kr + ((size_t)kv * S + (kb << 5)) * Dh;
        const ushort_t* Vgp = Vb + (size_t)(kv * (S / 32) + kb) * (Dh * 32);
        // each block is 8KB = 512 16B chunks; 4 per thread per tensor
#pragma unroll
        for (int i = 0; i < 4; ++i) {
            int e = tid + i * 128;
            unsigned klds = (unsigned)(size_t)&Kbuf[buf][e * 8];
            unsigned vlds = (unsigned)(size_t)&Vbuf[buf][e * 8];
            async_b128(klds, (unsigned)(e * 16), Kgp);
            async_b128(vlds, (unsigned)(e * 16), Vgp);
        }
    };

    // Q A-fragments (16 rows x 128 K), held in registers for the whole kernel
    Frag16 qa[4];
    {
        const ushort_t* qrow = &Qr[((size_t)h * S + qr0 + l16) * Dh];
#pragma unroll
        for (int t = 0; t < 4; ++t)
#pragma unroll
            for (int j = 0; j < 8; ++j) {
                int k = t * 32 + ((j < 4) ? 2 * j : 2 * j + 8) + 8 * half;
                qa[t].u[j] = *(const uint_t*)(qrow + k);
            }
    }

    float m[8], l[8];
    v8f zero = {};
    v8f o[8];
#pragma unroll
    for (int r = 0; r < 8; ++r) { m[r] = -3.0e38f; l[r] = 0.0f; }
#pragma unroll
    for (int n = 0; n < 8; ++n) o[n] = zero;

    const int kbEnd = (q0 + 64) >> 5;
    issueKV(0, 0);
    for (int kb = 0; kb < kbEnd; ++kb) {
        const int key0 = kb << 5;
        wait_async0();
        __syncthreads();
        if (kb + 1 < kbEnd) issueKV(kb + 1, (kb + 1) & 1);

        const uint_t* Ksu = (const uint_t*)Kbuf[kb & 1];
        const uint_t* Vtu = (const uint_t*)Vbuf[kb & 1];

        // ---- scores: S(16x32) = Q(16x128) * K^T, 8 WMMAs ----
        v8f sc[2]; sc[0] = zero; sc[1] = zero;
#pragma unroll
        for (int tn = 0; tn < 2; ++tn) {
            int col = tn * 16 + l16;
#pragma unroll
            for (int t = 0; t < 4; ++t) {
                Frag16 b;
#pragma unroll
                for (int j = 0; j < 8; ++j)
                    b.u[j] = Ksu[col * 64 + t * 16 + ((j < 4) ? j : j + 4) + 4 * half];
                sc[tn] = __builtin_amdgcn_wmma_f32_16x16x32_bf16(
                    false, qa[t].v, false, b.v, (short)0, sc[tn], false, false);
            }
        }

        // ---- causal mask + online softmax ----
        float p0[8], p1[8], corr[8];
#pragma unroll
        for (int r = 0; r < 8; ++r) {
            int qrow = qr0 + r + 8 * half;
            float s0 = (key0 + l16      <= qrow) ? sc[0][r] : -3.0e38f;
            float s1 = (key0 + 16 + l16 <= qrow) ? sc[1][r] : -3.0e38f;
            float rm = fmaxf(s0, s1);
            rm = fmaxf(rm, __shfl_xor(rm, 1, 32));
            rm = fmaxf(rm, __shfl_xor(rm, 2, 32));
            rm = fmaxf(rm, __shfl_xor(rm, 4, 32));
            rm = fmaxf(rm, __shfl_xor(rm, 8, 32));
            float mn = fmaxf(m[r], rm);
            corr[r] = expf(m[r] - mn);
            p0[r] = expf(s0 - mn);
            p1[r] = expf(s1 - mn);
            float rs = p0[r] + p1[r];
            rs += __shfl_xor(rs, 1, 32);
            rs += __shfl_xor(rs, 2, 32);
            rs += __shfl_xor(rs, 4, 32);
            rs += __shfl_xor(rs, 8, 32);
            l[r] = l[r] * corr[r] + rs;
            m[r] = mn;
        }
#pragma unroll
        for (int n = 0; n < 8; ++n)
#pragma unroll
            for (int r = 0; r < 8; ++r) o[n][r] *= corr[r];

        // ---- P (C layout) -> A layout via per-wave LDS bounce ----
#pragma unroll
        for (int r = 0; r < 8; ++r) {
            int mrow = r + 8 * half;
            Ps[wave][mrow * 32 + l16]      = f2bf1(p0[r]);
            Ps[wave][mrow * 32 + 16 + l16] = f2bf1(p1[r]);
        }
        __builtin_amdgcn_wave_barrier();
        Frag16 pa;
        uint_t* Psu = (uint_t*)Ps[wave];
#pragma unroll
        for (int j = 0; j < 8; ++j)
            pa.u[j] = Psu[l16 * 16 + ((j < 4) ? j : j + 4) + 4 * half];

        // ---- O(16x128) += P(16x32) * V(32x128), 8 WMMAs ----
#pragma unroll
        for (int n = 0; n < 8; ++n) {
            Frag16 b;
            int dcol = n * 16 + l16;
#pragma unroll
            for (int j = 0; j < 8; ++j)
                b.u[j] = Vtu[dcol * 16 + ((j < 4) ? j : j + 4) + 4 * half];
            o[n] = __builtin_amdgcn_wmma_f32_16x16x32_bf16(
                false, pa.v, false, b.v, (short)0, o[n], false, false);
        }
    }

    // ---- epilogue: normalize, bf16 store to AO[s][h*128+d] ----
#pragma unroll
    for (int r = 0; r < 8; ++r) {
        float inv = 1.0f / l[r];
        int row = qr0 + r + 8 * half;
        ushort_t* dst = &AO[(size_t)row * D + h * Dh];
#pragma unroll
        for (int n = 0; n < 8; ++n)
            dst[n * 16 + l16] = f2bf1(o[n][r] * inv);
    }
}

// ---------------------------------------------------------------------------
extern "C" void kernel_launch(void* const* d_in, const int* in_sizes, int n_in,
                              void* d_out, int out_size, void* d_ws, size_t ws_size,
                              hipStream_t stream) {
    (void)in_sizes; (void)n_in; (void)out_size; (void)ws_size;
    const float* x    = (const float*)d_in[0];
    const float* Wq   = (const float*)d_in[1];
    const float* Wk   = (const float*)d_in[2];
    const float* Wv   = (const float*)d_in[3];
    const float* Wo   = (const float*)d_in[4];
    const float* cosb = (const float*)d_in[5];
    const float* sinb = (const float*)d_in[6];
    float* out = (float*)d_out;

    char* w = (char*)d_ws;
    size_t off = 0;
    ushort_t* xb  = (ushort_t*)(w + off); off += (size_t)S * D * 2;
    ushort_t* Wqb = (ushort_t*)(w + off); off += (size_t)D * D * 2;
    ushort_t* Wkb = (ushort_t*)(w + off); off += (size_t)KV * Dh * D * 2;
    ushort_t* Wvb = (ushort_t*)(w + off); off += (size_t)KV * Dh * D * 2;
    ushort_t* Wob = (ushort_t*)(w + off); off += (size_t)D * D * 2;
    float*    Qf  = (float*)(w + off);    off += (size_t)S * D * 4;
    float*    Kf  = (float*)(w + off);    off += (size_t)S * KV * Dh * 4;
    float*    Vf  = (float*)(w + off);    off += (size_t)S * KV * Dh * 4;
    ushort_t* Qr  = (ushort_t*)(w + off); off += (size_t)H * S * Dh * 2;
    ushort_t* Kr  = (ushort_t*)(w + off); off += (size_t)KV * S * Dh * 2;
    ushort_t* Vb  = (ushort_t*)(w + off); off += (size_t)KV * S * Dh * 2;
    ushort_t* AO  = (ushort_t*)(w + off); off += (size_t)S * D * 2;

    // one-time bf16 conversion of activations + all weights
    auto cvt = [&](const float* s, ushort_t* d, size_t n) {
        int n8 = (int)(n / 8);
        f32_to_bf16<<<(n8 + 255) / 256, 256, 0, stream>>>(s, d, n8);
    };
    cvt(x,  xb,  (size_t)S * D);
    cvt(Wq, Wqb, (size_t)D * D);
    cvt(Wk, Wkb, (size_t)KV * Dh * D);
    cvt(Wv, Wvb, (size_t)KV * Dh * D);
    cvt(Wo, Wob, (size_t)D * D);

    // QKV projections (bf16 WMMA, async double-buffered LDS)
    gemm_wmma_async<<<dim3(D / 128, S / 128), 256, 0, stream>>>(xb, Wqb, Qf, S, D, D);
    gemm_wmma_async<<<dim3((KV * Dh) / 128, S / 128), 256, 0, stream>>>(xb, Wkb, Kf, S, KV * Dh, D);
    gemm_wmma_async<<<dim3((KV * Dh) / 128, S / 128), 256, 0, stream>>>(xb, Wvb, Vf, S, KV * Dh, D);
    // RoPE + bf16 repack (V block-transposed for async staging)
    size_t total = (size_t)H * S * 64 + (size_t)KV * S * 64 + (size_t)KV * S * Dh;
    rope_convert<<<(unsigned)((total + 255) / 256), 256, 0, stream>>>(Qf, Kf, Vf, cosb, sinb, Qr, Kr, Vb);
    // Flash attention (GQA, causal)
    flash_attn<<<dim3(S / 64, H), 128, 0, stream>>>(Qr, Kr, Vb, AO);
    // Output projection straight into d_out
    gemm_wmma_async<<<dim3(D / 128, S / 128), 256, 0, stream>>>(AO, Wob, out, S, D, D);
}
